// ShiftWindowMHA_12738873000253
// MI455X (gfx1250) — matrix operand order repaired and checked by reading, not compiled, not run on hardware
//
#include <hip/hip_runtime.h>

// ---------------------------------------------------------------------------
// CDNA5 / gfx1250 implementation of SwinV2 shifted-window MHA.
// Heavy math via v_wmma_f32_16x16x32_bf16 (wave32 WMMA); GEMM staging via
// async global->LDS DMA (ASYNCcnt) with double-buffered LDS.
// ---------------------------------------------------------------------------

typedef __attribute__((ext_vector_type(16))) __bf16 v16bf;
typedef __attribute__((ext_vector_type(8)))  float  v8f;

struct Frag32 { uint4 a, b; };   // 32 bytes == one v16bf fragment

__device__ __forceinline__ unsigned short f2bf(float f) {
  unsigned u = __builtin_bit_cast(unsigned, f);
  unsigned r = u + 0x7FFFu + ((u >> 16) & 1u);      // round-to-nearest-even
  return (unsigned short)(r >> 16);
}
__device__ __forceinline__ float bf2f(unsigned short b) {
  unsigned u = ((unsigned)b) << 16;
  return __builtin_bit_cast(float, u);
}

// Async global->LDS 16B copy (per lane). VDST = LDS byte offset, VADDR = u64.
__device__ __forceinline__ void async_b128(unsigned lds_off,
                                           const void* gaddr) {
  asm volatile("global_load_async_to_lds_b128 %0, %1, off"
               :: "v"(lds_off), "v"((unsigned long long)gaddr)
               : "memory");
}
__device__ __forceinline__ void wait_async() {
  asm volatile("s_wait_asynccnt 0" ::: "memory");
}
__device__ __forceinline__ unsigned lds_off_of(const void* p) {
  return (unsigned)(unsigned long long)p;   // low 32 bits = LDS offset
}

// A-matrix fragment (16x32 bf16), row-major source with leading dim `ld`.
// ISA layout: lanes 0-15 row M=l, K pairs {0..7,16..23}; lanes 16-31 K+8.
__device__ __forceinline__ v16bf load_afrag(const unsigned short* base, int ld,
                                            int row0, int k0, int lane) {
  int r = row0 + (lane & 15);
  int k = k0 + ((lane >> 4) << 3);
  Frag32 f;
  f.a = *(const uint4*)(base + r * ld + k);
  f.b = *(const uint4*)(base + r * ld + k + 16);
  return __builtin_bit_cast(v16bf, f);
}
// B-matrix fragment (32x16 bf16): column n sourced from row n of `base`.
// lanes 0-15: K {0..7,16..23}, lanes 16-31: K+8 (matches A).
__device__ __forceinline__ v16bf load_bfrag(const unsigned short* base, int ld,
                                            int col0, int k0, int lane) {
  int r = col0 + (lane & 15);
  int k = k0 + ((lane >> 4) << 3);
  Frag32 f;
  f.a = *(const uint4*)(base + r * ld + k);
  f.b = *(const uint4*)(base + r * ld + k + 16);
  return __builtin_bit_cast(v16bf, f);
}

__device__ __forceinline__ v8f wmma_bf16(v16bf a, v16bf b, v8f c) {
  return __builtin_amdgcn_wmma_f32_16x16x32_bf16(false, a, false, b,
                                                 (short)0, c, false, false);
}

// ---------------------------------------------------------------------------
// fp32 -> bf16 bulk convert (weights)
// ---------------------------------------------------------------------------
__global__ __launch_bounds__(256) void k_conv_bf16(const float* __restrict__ s,
                                                   unsigned short* __restrict__ d,
                                                   int n4) {
  int i = blockIdx.x * 256 + threadIdx.x;
  if (i >= n4) return;
  float4 f = ((const float4*)s)[i];
  unsigned lo = (unsigned)f2bf(f.x) | ((unsigned)f2bf(f.y) << 16);
  unsigned hi = (unsigned)f2bf(f.z) | ((unsigned)f2bf(f.w) << 16);
  ((uint2*)d)[i] = make_uint2(lo, hi);
}

// ---------------------------------------------------------------------------
// x [128,16,16,1024] f32 --(roll -4,-4 + window partition)--> xw [32768,1024] bf16
// ---------------------------------------------------------------------------
__global__ __launch_bounds__(256) void k_prep_x(const float* __restrict__ x,
                                                unsigned short* __restrict__ xw) {
  int row = blockIdx.x;                 // 0..32767
  int w = row >> 6, n = row & 63;
  int b = w >> 2, wh = (w >> 1) & 1, ww = w & 1;
  int ih = n >> 3, iw = n & 7;
  int sh = (wh * 8 + ih + 4) & 15;      // shifted source coords
  int sw = (ww * 8 + iw + 4) & 15;
  const float* src = x + (((size_t)b * 16 + sh) * 16 + sw) * 1024;
  unsigned short* dst = xw + (size_t)row * 1024;
  int c = threadIdx.x * 4;
  float4 f = *(const float4*)(src + c);
  unsigned lo = (unsigned)f2bf(f.x) | ((unsigned)f2bf(f.y) << 16);
  unsigned hi = (unsigned)f2bf(f.z) | ((unsigned)f2bf(f.w) << 16);
  *(uint2*)(dst + c) = make_uint2(lo, hi);
}

// ---------------------------------------------------------------------------
// CPB MLP: table entry e=(i*15+j) -> 16 head biases. 225 blocks x 64 threads.
// ---------------------------------------------------------------------------
__device__ __forceinline__ float cpb_coord(int i) {
  float v = (float)(i - 7) * (8.0f / 7.0f);
  float a = fabsf(v);
  float r = log2f(a + 1.0f) * (1.0f / 3.0f);
  return (v < 0.0f) ? -r : r;
}

__global__ __launch_bounds__(64) void k_cpb_tab(const float* __restrict__ w1,
                                                const float* __restrict__ b1,
                                                const float* __restrict__ w2,
                                                float* __restrict__ tab) {
  __shared__ float hid[512];
  int e = blockIdx.x;
  int i = e / 15, j = e % 15;
  float c0 = cpb_coord(i), c1 = cpb_coord(j);
  for (int u = threadIdx.x; u < 512; u += 64)
    hid[u] = fmaxf(w1[u * 2] * c0 + w1[u * 2 + 1] * c1 + b1[u], 0.0f);
  __syncthreads();
  if (threadIdx.x < 16) {
    int hd = threadIdx.x;
    float s = 0.0f;
    for (int u = 0; u < 512; ++u) s += w2[hd * 512 + u] * hid[u];
    tab[e * 16 + hd] = s;
  }
}

// ---------------------------------------------------------------------------
// Badd[wi][h][q][k] = 16*sigmoid(cpb) ; masked pairs -> -1e30
// ---------------------------------------------------------------------------
__device__ __forceinline__ int regid(int x) { return x < 8 ? 0 : (x < 12 ? 1 : 2); }

__global__ __launch_bounds__(256) void k_badd(const float* __restrict__ tab,
                                              float* __restrict__ Badd) {
  int blk = blockIdx.x;                 // wi*16 + h  (64 blocks)
  int wi = blk >> 4, h = blk & 15;
  int wh = wi >> 1, ww = wi & 1;
  for (int p = threadIdx.x; p < 4096; p += 256) {
    int qi = p >> 6, ki = p & 63;
    int qh = qi >> 3, qw = qi & 7, kh = ki >> 3, kw = ki & 7;
    int ridx = (qh - kh + 7) * 15 + (qw - kw + 7);
    float b = tab[ridx * 16 + h];
    float bias = 16.0f / (1.0f + __expf(-b));
    int rq = regid(wh * 8 + qh) * 3 + regid(ww * 8 + qw);
    int rk = regid(wh * 8 + kh) * 3 + regid(ww * 8 + kw);
    if (rq != rk) bias = -1e30f;
    Badd[(size_t)blk * 4096 + p] = bias;
  }
}

// ---------------------------------------------------------------------------
// WMMA GEMM:  C[M=32768, N=1024] = A[M,1024] * W[N,1024]^T
// block: 256 thr (8 waves) -> 128x128 tile; wave -> 32x64; K-step 32.
// Double-buffered LDS filled by async global->LDS DMA (ASYNCcnt).
// mode 0: bf16 row-major out (Q,K)
// mode 1: +bias, V transposed-per-head out  vT[w][h][d][n]
// mode 2: +bias, f32 scatter to d_out with window-merge + reverse shift
// ---------------------------------------------------------------------------
__global__ __launch_bounds__(256) void k_gemm(const unsigned short* __restrict__ A,
                                              const unsigned short* __restrict__ W,
                                              const float* __restrict__ bias,
                                              unsigned short* __restrict__ outb,
                                              float* __restrict__ outf,
                                              int mode) {
  __shared__ unsigned short As[2][128][40];   // 32 bf16 + pad -> 80B rows
  __shared__ unsigned short Ws[2][128][40];   // total 40 KB
  const int t = threadIdx.x, lane = t & 31, wave = t >> 5;
  const int m0 = blockIdx.x * 128, n0 = blockIdx.y * 128;
  const int wm = (wave & 3) * 32, wn = (wave >> 2) * 64;

  v8f acc[2][4];
  for (int a = 0; a < 2; ++a)
    for (int b = 0; b < 4; ++b)
      for (int i = 0; i < 8; ++i) acc[a][b][i] = 0.0f;

  // per-thread staging slot: row rA, 16-element half `halfk`
  const int rA = t >> 1, halfk = t & 1;
  const unsigned short* Ag = A + (size_t)(m0 + rA) * 1024 + halfk * 16;
  const unsigned short* Wg = W + (size_t)(n0 + rA) * 1024 + halfk * 16;

  auto issue_tile = [&](int buf, int k0) {
    unsigned la = lds_off_of(&As[buf][rA][halfk * 16]);
    unsigned lw = lds_off_of(&Ws[buf][rA][halfk * 16]);
    async_b128(la,      Ag + k0);
    async_b128(la + 16, Ag + k0 + 8);
    async_b128(lw,      Wg + k0);
    async_b128(lw + 16, Wg + k0 + 8);
  };

  issue_tile(0, 0);
  wait_async();
  __syncthreads();

  for (int ks = 0; ks < 32; ++ks) {
    const int cur = ks & 1;
    if (ks < 31) issue_tile(cur ^ 1, (ks + 1) * 32);   // overlap with WMMAs

    v16bf af[2];
#pragma unroll
    for (int mi = 0; mi < 2; ++mi)
      af[mi] = load_afrag(&As[cur][0][0], 40, wm + mi * 16, 0, lane);
#pragma unroll
    for (int ni = 0; ni < 4; ++ni) {
      v16bf b = load_bfrag(&Ws[cur][0][0], 40, wn + ni * 16, 0, lane);
#pragma unroll
      for (int mi = 0; mi < 2; ++mi)
        acc[mi][ni] = wmma_bf16(af[mi], b, acc[mi][ni]);
    }
    wait_async();          // next buffer fully in LDS
    __syncthreads();
  }

  const int hl = lane >> 4, nn = lane & 15;
#pragma unroll
  for (int mi = 0; mi < 2; ++mi)
#pragma unroll
    for (int ni = 0; ni < 4; ++ni)
#pragma unroll
      for (int i = 0; i < 8; ++i) {
        int m = m0 + wm + mi * 16 + i + hl * 8;
        int n = n0 + wn + ni * 16 + nn;
        float v = acc[mi][ni][i];
        if (mode == 0) {
          outb[(size_t)m * 1024 + n] = f2bf(v);
        } else if (mode == 1) {
          v += bias[n];
          int w = m >> 6, ntok = m & 63, hh = n >> 6, dd = n & 63;
          outb[(((size_t)w * 16 + hh) * 64 + dd) * 64 + ntok] = f2bf(v);
        } else {
          v += bias[n];
          int w = m >> 6, ntok = m & 63;
          int b = w >> 2, wh = (w >> 1) & 1, ww = w & 1;
          int ih = ntok >> 3, iw = ntok & 7;
          int oh = (wh * 8 + ih + 4) & 15;
          int ow = (ww * 8 + iw + 4) & 15;
          outf[(((size_t)b * 16 + oh) * 16 + ow) * 1024 + n] = v;
        }
      }
}

// ---------------------------------------------------------------------------
// Fused cosine attention: one wave per (window, head). 2 waves / block.
// S = qn*kn^T * exp(min(ls,log100)) + Badd ; softmax ; O = P*V
// ---------------------------------------------------------------------------
__global__ __launch_bounds__(64) void k_attn(const unsigned short* __restrict__ Q,
                                             const unsigned short* __restrict__ K,
                                             const unsigned short* __restrict__ VT,
                                             const float* __restrict__ Badd,
                                             const float* __restrict__ lscale,
                                             unsigned short* __restrict__ O) {
  __shared__ unsigned short qs[2][64 * 64];
  __shared__ unsigned short ks[2][64 * 64];
  __shared__ unsigned short vs[2][64 * 64];
  __shared__ unsigned short pb[2][16 * 64];
  __shared__ float          ps[2][16 * 64];

  const int t = threadIdx.x, lane = t & 31, wv = t >> 5;
  const int pair = blockIdx.x * 2 + wv;         // 0..8191
  const int w = pair >> 4, h = pair & 15, wi = w & 3;

  unsigned short* q_l = qs[wv];
  unsigned short* k_l = ks[wv];
  unsigned short* v_l = vs[wv];
  unsigned short* p_l = pb[wv];
  float*          s_l = ps[wv];

  const float scale = __expf(fminf(lscale[h], 4.6051702f));

  // --- async-copy V^T slab [64 d][64 n] for this head (overlaps q/k work) ---
  {
    const unsigned short* src = VT + (((size_t)w * 16 + h) * 64) * 64;
#pragma unroll
    for (int j = 0; j < 16; ++j) {
      int idx = lane + j * 32;          // 512 x b128
      async_b128(lds_off_of(&v_l[idx * 8]), src + idx * 8);
    }
  }

  // --- load + row-normalize q and k into LDS (2 rows per lane) ---
  const unsigned short* qbase = Q + ((size_t)w * 64) * 1024 + h * 64;
  const unsigned short* kbase = K + ((size_t)w * 64) * 1024 + h * 64;
  for (int which = 0; which < 2; ++which) {
    const unsigned short* gb = which ? kbase : qbase;
    unsigned short* dst = which ? k_l : q_l;
#pragma unroll
    for (int rr = 0; rr < 2; ++rr) {
      int r = lane * 2 + rr;
      const unsigned short* src = gb + (size_t)r * 1024;
      uint4 buf[8];
      float ss = 0.0f;
#pragma unroll
      for (int j = 0; j < 8; ++j) {
        buf[j] = *(const uint4*)(src + j * 8);
        const unsigned short* pu = (const unsigned short*)&buf[j];
#pragma unroll
        for (int e = 0; e < 8; ++e) { float f = bf2f(pu[e]); ss += f * f; }
      }
      float inv = 1.0f / fmaxf(sqrtf(ss), 1e-12f);
#pragma unroll
      for (int j = 0; j < 8; ++j) {
        const unsigned short* pu = (const unsigned short*)&buf[j];
        union { unsigned short u[8]; uint4 q; } o;
#pragma unroll
        for (int e = 0; e < 8; ++e) o.u[e] = f2bf(bf2f(pu[e]) * inv);
        *(uint4*)&dst[r * 64 + j * 8] = o.q;
      }
    }
  }
  wait_async();
  __syncthreads();

  const float* bb = Badd + ((size_t)wi * 16 + h) * 4096;
  const int hl = lane >> 4, nn = lane & 15;

  for (int mt = 0; mt < 4; ++mt) {
    // ---- S block [16 x 64] ----
    v8f sacc[4];
    for (int nt = 0; nt < 4; ++nt)
      for (int i = 0; i < 8; ++i) sacc[nt][i] = 0.0f;
#pragma unroll
    for (int kt = 0; kt < 2; ++kt) {
      v16bf a = load_afrag(q_l, 64, mt * 16, kt * 32, lane);
#pragma unroll
      for (int nt = 0; nt < 4; ++nt) {
        v16bf b = load_bfrag(k_l, 64, nt * 16, kt * 32, lane);
        sacc[nt] = wmma_bf16(a, b, sacc[nt]);
      }
    }
    // scale + bias(+mask), spill to LDS
#pragma unroll
    for (int nt = 0; nt < 4; ++nt)
#pragma unroll
      for (int i = 0; i < 8; ++i) {
        int ml = i + hl * 8;                     // local query row 0..15
        int kc = nt * 16 + nn;                   // key col 0..63
        float v = sacc[nt][i] * scale + bb[(size_t)(mt * 16 + ml) * 64 + kc];
        s_l[ml * 64 + kc] = v;
      }
    __syncthreads();
    // ---- row softmax (lanes 0..15, one row each) ----
    if (lane < 16) {
      int r = lane;
      float mx = -3.0e38f;
      for (int c = 0; c < 64; ++c) mx = fmaxf(mx, s_l[r * 64 + c]);
      float sum = 0.0f;
      for (int c = 0; c < 64; ++c) {
        float e = __expf(s_l[r * 64 + c] - mx);
        s_l[r * 64 + c] = e;
        sum += e;
      }
      float rinv = 1.0f / sum;
      for (int c = 0; c < 64; ++c) p_l[r * 64 + c] = f2bf(s_l[r * 64 + c] * rinv);
    }
    __syncthreads();
    // ---- O block [16 x 64] = P x V ----
#pragma unroll
    for (int dt = 0; dt < 4; ++dt) {
      v8f oacc;
      for (int i = 0; i < 8; ++i) oacc[i] = 0.0f;
#pragma unroll
      for (int kt = 0; kt < 2; ++kt) {
        v16bf a = load_afrag(p_l, 64, 0, kt * 32, lane);
        v16bf b = load_bfrag(v_l, 64, dt * 16, kt * 32, lane);
        oacc = wmma_bf16(a, b, oacc);
      }
#pragma unroll
      for (int i = 0; i < 8; ++i) {
        int grow = w * 64 + mt * 16 + i + hl * 8;
        int gcol = h * 64 + dt * 16 + nn;
        O[(size_t)grow * 1024 + gcol] = f2bf(oacc[i]);
      }
    }
    __syncthreads();
  }
}

// ---------------------------------------------------------------------------
// Host launcher
// ---------------------------------------------------------------------------
extern "C" void kernel_launch(void* const* d_in, const int* in_sizes, int n_in,
                              void* d_out, int out_size, void* d_ws, size_t ws_size,
                              hipStream_t stream) {
  (void)in_sizes; (void)n_in; (void)out_size; (void)ws_size;
  const float* x   = (const float*)d_in[0];
  const float* Wq  = (const float*)d_in[1];
  const float* Wk  = (const float*)d_in[2];
  const float* Wv  = (const float*)d_in[3];
  const float* bv  = (const float*)d_in[4];
  const float* Wo  = (const float*)d_in[5];
  const float* bo  = (const float*)d_in[6];
  const float* ls  = (const float*)d_in[7];
  const float* w1  = (const float*)d_in[8];
  const float* b1  = (const float*)d_in[9];
  const float* w2  = (const float*)d_in[10];
  float* out = (float*)d_out;

  char* ws = (char*)d_ws;
  size_t off = 0;
  auto alloc = [&](size_t bytes) -> void* {
    void* p = ws + off;
    off += (bytes + 255) & ~(size_t)255;
    return p;
  };
  const size_t ACT = (size_t)32768 * 1024 * 2;   // 64 MB bf16
  unsigned short* xw  = (unsigned short*)alloc(ACT);   // also reused as O
  unsigned short* qb  = (unsigned short*)alloc(ACT);
  unsigned short* kb  = (unsigned short*)alloc(ACT);
  unsigned short* vtb = (unsigned short*)alloc(ACT);
  unsigned short* wqb = (unsigned short*)alloc((size_t)1024 * 1024 * 2);
  unsigned short* wkb = (unsigned short*)alloc((size_t)1024 * 1024 * 2);
  unsigned short* wvb = (unsigned short*)alloc((size_t)1024 * 1024 * 2);
  unsigned short* wob = (unsigned short*)alloc((size_t)1024 * 1024 * 2);
  float* tab  = (float*)alloc((size_t)225 * 16 * 4);
  float* badd = (float*)alloc((size_t)64 * 4096 * 4);

  const int W4 = (1024 * 1024) / 4;   // float4 chunks per weight matrix
  k_conv_bf16<<<(W4 + 255) / 256, 256, 0, stream>>>(Wq, wqb, W4);
  k_conv_bf16<<<(W4 + 255) / 256, 256, 0, stream>>>(Wk, wkb, W4);
  k_conv_bf16<<<(W4 + 255) / 256, 256, 0, stream>>>(Wv, wvb, W4);
  k_conv_bf16<<<(W4 + 255) / 256, 256, 0, stream>>>(Wo, wob, W4);

  k_prep_x<<<32768, 256, 0, stream>>>(x, xw);
  k_cpb_tab<<<225, 64, 0, stream>>>(w1, b1, w2, tab);
  k_badd<<<64, 256, 0, stream>>>(tab, badd);

  dim3 ggrid(256, 8);   // M/128 x N/128
  k_gemm<<<ggrid, 256, 0, stream>>>(xw, wqb, nullptr, qb, nullptr, 0);
  k_gemm<<<ggrid, 256, 0, stream>>>(xw, wkb, nullptr, kb, nullptr, 0);
  k_gemm<<<ggrid, 256, 0, stream>>>(xw, wvb, bv, vtb, nullptr, 1);

  k_attn<<<4096, 64, 0, stream>>>(qb, kb, vtb, badd, ls, xw);  // O -> xw

  k_gemm<<<ggrid, 256, 0, stream>>>(xw, wob, bo, nullptr, out, 2);
}
